// interskeleton_interaction_S2_67568425501193
// MI455X (gfx1250) — compile-verified, matrix operand order checked by reading, not run on hardware
//
#include <hip/hip_runtime.h>
#include <hip/hip_bf16.h>

typedef __attribute__((ext_vector_type(16))) __bf16 v16bf;
typedef __attribute__((ext_vector_type(8)))  float  v8f;

constexpr int DH   = 120;    // hidden
constexpr int DHP  = 128;    // hidden padded
constexpr int DIN  = 2560;   // input feature
constexpr int NV   = 25;     // joints
constexpr int NBATCH = 1024;
constexpr int NB2  = 512;
constexpr int ROWS = NBATCH * NV;   // 25600

__device__ __forceinline__ unsigned short f2bf(float f) {
  unsigned u = __float_as_uint(f);
  u += 0x7FFFu + ((u >> 16) & 1u);          // round-to-nearest-even
  return (unsigned short)(u >> 16);
}

// Swizzle element (m,k) of a 16x32 bf16 WMMA operand tile into the per-lane
// fragment layout (ISA 7.12.2): lane = m + 16*k[3], elem = 8*(k>>4) | k&7.
__device__ __forceinline__ int frag_pos(int m, int k) {
  int lane = m + (((k >> 3) & 1) << 4);
  int j    = ((k >> 4) << 3) | (k & 7);
  return lane * 16 + j;
}

__device__ __forceinline__ v16bf frag_ld(const unsigned short* buf, int lane) {
  return *(const v16bf*)(buf + lane * 16);   // contiguous 32B per lane
}

__device__ __forceinline__ v8f wmma_bf16(v16bf a, v16bf b, v8f c) {
  return __builtin_amdgcn_wmma_f32_16x16x32_bf16(false, a, false, b, (short)0, c,
                                                 false, false);
}

// Async global->LDS DMA (16B per lane), tracked by ASYNCcnt (ISA 10.x).
__device__ __forceinline__ void async_cp16(unsigned lds_dst, const void* gsrc) {
  asm volatile("global_load_async_to_lds_b128 %0, %1, off"
               :: "v"(lds_dst), "v"((unsigned long long)(size_t)gsrc)
               : "memory");
}
__device__ __forceinline__ void wait_async() {
  asm volatile("s_wait_asynccnt 0" ::: "memory");
}
__device__ __forceinline__ unsigned lds_off32(const void* p) {
  return (unsigned)(size_t)p;   // generic LDS addr: low 32 bits = LDS offset
}

// Swizzled global index for y (A-operand tiles of GEMM2): contiguous 32KB per
// 128-row block, ordered (mtile, ktile, fragpos).
__device__ __forceinline__ size_t y_idx(int r, int c) {
  int mb = r >> 7, mt = (r >> 4) & 7, m = r & 15;
  return ((size_t)(mb * 8 + mt) * 4 + (c >> 5)) * 512 + frag_pos(m, c & 31);
}

// ---------------------------------------------------------------- prep ------
// All weight operands stored pre-swizzled in WMMA fragment-tile order so the
// GEMM kernels can DMA them linearly into LDS.
__global__ __launch_bounds__(256) void k_prep(
    const float* __restrict__ fc_w, const float* __restrict__ fc2_w,
    const float* __restrict__ w12,  const float* __restrict__ w21,
    unsigned short* __restrict__ Bw1s, unsigned short* __restrict__ Bw2s,
    unsigned short* __restrict__ B12s, unsigned short* __restrict__ B21s) {
  int idx = blockIdx.x * 256 + threadIdx.x;
  if (idx < DIN * DHP) {                       // Bw1s: per-kchunk (nt, frag)
    int k = idx / DHP, n = idx % DHP;
    unsigned short v = (n < DH) ? f2bf(fc_w[(size_t)n * DIN + k]) : (unsigned short)0;
    Bw1s[(size_t)((k >> 5) * 8 + (n >> 4)) * 512 + frag_pos(n & 15, k & 31)] = v;
  }
  int i2 = idx - DIN * DHP;
  if (i2 >= 0 && i2 < DHP * DIN) {             // Bw2s: per-nblock (nt, kt, frag)
    int k = i2 / DIN, n = i2 % DIN;
    unsigned short v = (k < DH) ? f2bf(fc2_w[(size_t)n * DH + k]) : (unsigned short)0;
    Bw2s[((size_t)((n >> 7) * 8 + ((n >> 4) & 7)) * 4 + (k >> 5)) * 512 +
         frag_pos(n & 15, k & 31)] = v;
  }
  int i3 = idx - 2 * DIN * DHP;
  if (i3 >= 0 && i3 < DHP * DHP) {             // B12s/B21s: (nt, kt, frag)
    int k = i3 / DHP, n = i3 % DHP;
    bool ok = (k < DH) && (n < DH);
    size_t p = (size_t)((n >> 4) * 4 + (k >> 5)) * 512 + frag_pos(n & 15, k & 31);
    B12s[p] = ok ? f2bf(w12[n * DH + k]) : (unsigned short)0;
    B21s[p] = ok ? f2bf(w21[n * DH + k]) : (unsigned short)0;
  }
}

// ---------------------------------------------------------- GEMM1 -----------
// att_bf[r][n] = bf16( x_att[r][:] . fc_w[n][:] + fc_b[n] )
// A: f32->bf16 VALU staging; B: async DMA of pre-swizzled 8KB chunk.
__global__ __launch_bounds__(256) void k_gemm1(
    const float* __restrict__ xatt, const unsigned short* __restrict__ Bw1s,
    const float* __restrict__ fc_b, unsigned short* __restrict__ att_bf) {
  __shared__ __align__(32) unsigned short As[8 * 512];
  __shared__ __align__(32) unsigned short Bs[8 * 512];
  const int tid = threadIdx.x, wave = tid >> 5, lane = tid & 31;
  const int m0 = blockIdx.x * 128;
  const unsigned bs_base = lds_off32(Bs);
  v8f z = {};
  v8f acc[8];
  for (int i = 0; i < 8; ++i) acc[i] = z;

  for (int kk = 0; kk < DIN; kk += 32) {
    { // async copy B chunk (8KB, already fragment-ordered)
      const char* src = (const char*)(Bw1s + (size_t)(kk >> 5) * (8 * 512));
      int off = tid * 32;
      async_cp16(bs_base + off,      src + off);
      async_cp16(bs_base + off + 16, src + off + 16);
    }
    if (kk + 32 < DIN) {  // prefetch next A chunk -> global_prefetch_b8
      __builtin_prefetch(&xatt[(size_t)(m0 + (tid >> 1)) * DIN + kk + 32 +
                               ((tid & 1) << 4)], 0, 1);
    }
    for (int t = tid; t < 128 * 32; t += 256) {        // A: 128x32 f32->bf16
      int m = t >> 5, k = t & 31;
      As[(m >> 4) * 512 + frag_pos(m & 15, k)] =
          f2bf(xatt[(size_t)(m0 + m) * DIN + kk + k]);
    }
    wait_async();
    __syncthreads();
    v16bf a = frag_ld(As + wave * 512, lane);
    for (int nt = 0; nt < 8; ++nt)
      acc[nt] = wmma_bf16(a, frag_ld(Bs + nt * 512, lane), acc[nt]);
    __syncthreads();
  }
  int rbase = m0 + wave * 16 + ((lane >> 4) << 3);
  int c0 = lane & 15;
  for (int nt = 0; nt < 8; ++nt) {
    int col = nt * 16 + c0;
    float bias = (col < DH) ? fc_b[col] : 0.f;
    for (int j = 0; j < 8; ++j)
      att_bf[(size_t)(rbase + j) * DHP + col] = f2bf(acc[nt][j] + bias);
  }
}

// ---------------------------------------------------------- c = a @ W^T -----
// One WG per n in [0,1024): c[n] (25x120) = att[n] @ Wsel^T.  B via async DMA.
// Output c stored in B-fragment layout (8KB per batch) for k_attn DMA.
__global__ __launch_bounds__(256) void k_gemmc(
    const unsigned short* __restrict__ att_bf,
    const unsigned short* __restrict__ B12s,
    const unsigned short* __restrict__ B21s,
    unsigned short* __restrict__ c_bfs) {
  __shared__ __align__(32) unsigned short As[2 * 4 * 512];
  __shared__ __align__(32) unsigned short Bs[8 * 4 * 512];
  const int n = blockIdx.x;
  const unsigned short* Bg = (n & 1) ? B21s : B12s;
  const int tid = threadIdx.x, wave = tid >> 5, lane = tid & 31;
  const unsigned bs_base = lds_off32(Bs);

  for (int q = 0; q < 8; ++q) {                        // async B: 32KB linear
    int off = (q * 256 + tid) * 16;
    async_cp16(bs_base + off, (const char*)Bg + off);
  }
  for (int t = tid; t < 32 * 128; t += 256) {          // A: 25x120 pad 32x128
    int m = t >> 7, k = t & 127;
    As[((m >> 4) * 4 + (k >> 5)) * 512 + frag_pos(m & 15, k & 31)] =
        (m < NV) ? att_bf[(size_t)(n * NV + m) * DHP + k] : (unsigned short)0;
  }
  wait_async();
  __syncthreads();
  v8f acc0 = {}, acc1 = {};
  for (int kt = 0; kt < 4; ++kt) {
    v16bf b = frag_ld(Bs + (wave * 4 + kt) * 512, lane);
    acc0 = wmma_bf16(frag_ld(As + kt * 512, lane), b, acc0);
    acc1 = wmma_bf16(frag_ld(As + (4 + kt) * 512, lane), b, acc1);
  }
  size_t cbase = (size_t)n * 4096;
  int col = wave * 16 + (lane & 15);
  int rsub = (lane >> 4) << 3;
  for (int j = 0; j < 8; ++j) {
    int r0 = rsub + j;
    if (r0 < NV)
      c_bfs[cbase + ((r0 >> 4) * 4 + (col >> 5)) * 512 +
            frag_pos(r0 & 15, col & 31)] = f2bf(acc0[j]);
    int r1 = 16 + rsub + j;
    if (r1 < NV)
      c_bfs[cbase + ((r1 >> 4) * 4 + (col >> 5)) * 512 +
            frag_pos(r1 & 15, col & 31)] = f2bf(acc1[j]);
  }
  for (int t = tid; t < 32 * 128; t += 256) {          // zero pad rows 25..31
    int m = t >> 7, k = t & 127;
    if (m >= NV)
      c_bfs[cbase + ((m >> 4) * 4 + (k >> 5)) * 512 +
            frag_pos(m & 15, k & 31)] = 0;
  }
}

// ------------------------------------------------- fused attention ----------
__global__ __launch_bounds__(256) void k_attn(
    const unsigned short* __restrict__ att_bf,
    const unsigned short* __restrict__ c_bfs,
    const float* __restrict__ x_s1, unsigned short* __restrict__ y_bfs) {
  __shared__ __align__(32) unsigned short Aa1[2 * 4 * 512];
  __shared__ __align__(32) unsigned short Aa2[2 * 4 * 512];
  __shared__ __align__(32) unsigned short Bc1[2 * 4 * 512];
  __shared__ __align__(32) unsigned short Bc2[2 * 4 * 512];
  __shared__ __align__(32) unsigned short P12[2 * 512];
  __shared__ __align__(32) unsigned short P21[2 * 512];
  __shared__ __align__(32) unsigned short Bx1[8 * 512];
  __shared__ __align__(32) unsigned short Bx2[8 * 512];
  __shared__ float Lg[2 * 32 * 32];
  const int tid = threadIdx.x, wave = tid >> 5, lane = tid & 31;
  const int i = blockIdx.x, n1 = 2 * i, n2r = 2 * i + 1;

  // phase 1: async DMA c1/c2 (B-frag layout, 8KB each); stage a via VALU
  {
    const char* c1 = (const char*)(c_bfs + (size_t)n1 * 4096);
    const char* c2 = (const char*)(c_bfs + (size_t)n2r * 4096);
    unsigned b1 = lds_off32(Bc1), b2 = lds_off32(Bc2);
    int off = tid * 32;
    async_cp16(b1 + off, c1 + off); async_cp16(b1 + off + 16, c1 + off + 16);
    async_cp16(b2 + off, c2 + off); async_cp16(b2 + off + 16, c2 + off + 16);
  }
  for (int t = tid; t < 32 * 128; t += 256) {
    int m = t >> 7, k = t & 127;
    int p = ((m >> 4) * 4 + (k >> 5)) * 512 + frag_pos(m & 15, k & 31);
    Aa1[p] = (m < NV) ? att_bf[(size_t)(n1 * NV + m) * DHP + k] : (unsigned short)0;
    Aa2[p] = (m < NV) ? att_bf[(size_t)(n2r * NV + m) * DHP + k] : (unsigned short)0;
  }
  for (int t = tid; t < 1024; t += 256) { P12[t] = 0; P21[t] = 0; }
  wait_async();
  __syncthreads();

  // phase 2: logits via WMMA (8 waves = 2 matrices x 2mt x 2nt)
  {
    int which = wave >> 2, sub = wave & 3, mt = sub >> 1, nt = sub & 1;
    const unsigned short* A = which ? Aa1 : Aa2;
    const unsigned short* B = which ? Bc2 : Bc1;
    v8f acc = {};
    for (int kt = 0; kt < 4; ++kt)
      acc = wmma_bf16(frag_ld(A + (mt * 4 + kt) * 512, lane),
                      frag_ld(B + (nt * 4 + kt) * 512, lane), acc);
    int row = mt * 16 + ((lane >> 4) << 3), col = nt * 16 + (lane & 15);
    for (int j = 0; j < 8; ++j) Lg[which * 1024 + (row + j) * 32 + col] = acc[j];
  }
  // phase 2b: stage x1/x2 as B-frags (k = joint v, n = channel c = t*8+d)
  for (int t = tid; t < 32 * 128; t += 256) {
    int v = t >> 7, c = t & 127;
    int p = (c >> 4) * 512 + frag_pos(c & 15, v);
    float f1 = 0.f, f2 = 0.f;
    if (v < NV && c < DH) {
      size_t off = (size_t)(c & 7) * 375 + (size_t)(c >> 3) * 25 + v;
      f1 = x_s1[(size_t)n1 * 3000 + off];
      f2 = x_s1[(size_t)n2r * 3000 + off];
    }
    Bx1[p] = f2bf(f1); Bx2[p] = f2bf(f2);
  }
  __syncthreads();

  // phase 3: row softmax over 25 cols, write probs as A-frags; zero y pads
  if (tid < 50) {
    int which = tid / 25, row = tid % 25;
    float* L = &Lg[which * 1024 + row * 32];
    float mx = -1e30f;
    for (int c = 0; c < NV; ++c) mx = fmaxf(mx, L[c]);
    float s = 0.f;
    for (int c = 0; c < NV; ++c) { float e = __expf(L[c] - mx); L[c] = e; s += e; }
    float inv = 1.f / s;
    unsigned short* P = which ? P21 : P12;
    for (int c = 0; c < 32; ++c)
      P[(row >> 4) * 512 + frag_pos(row & 15, c)] =
          (c < NV) ? f2bf(L[c] * inv) : (unsigned short)0;
    int yrow = (which ? (NB2 + i) : i) * NV + row;
    for (int p2 = DH; p2 < DHP; ++p2) y_bfs[y_idx(yrow, p2)] = 0;
  }
  __syncthreads();

  // phase 4: x_after = P @ x + residual; channel-permute into swizzled y
  for (int q = 0; q < 4; ++q) {
    int t = wave * 4 + q;
    int which = t >> 4, rem = t & 15, mt = rem >> 3, nt = rem & 7;
    const unsigned short* A = which ? P21 : P12;
    const unsigned short* B = which ? Bx2 : Bx1;
    v8f acc = {};
    acc = wmma_bf16(frag_ld(A + mt * 512, lane), frag_ld(B + nt * 512, lane), acc);
    int rsub = (lane >> 4) << 3;
    int col = nt * 16 + (lane & 15);
    if (col < DH) {
      int d = col & 7, tt = col >> 3;
      int pc = (col % 15) * 8 + col / 15;                 // reshape permutation
      int nres = which ? n1 : n2r;
      int ybase = (which ? (NB2 + i) : i) * NV;
      for (int j = 0; j < 8; ++j) {
        int row = mt * 16 + rsub + j;
        if (row < NV) {
          float res = x_s1[(size_t)nres * 3000 + (size_t)d * 375 + tt * 25 + row];
          y_bfs[y_idx(ybase + row, pc)] = f2bf(acc[j] + res);
        }
      }
    }
  }
}

// ---------------------------------------------------------- GEMM2 -----------
// out = y @ fc2_w^T + fc2_b. Both operands pre-swizzled -> pure async DMA.
__global__ __launch_bounds__(256) void k_gemm2(
    const unsigned short* __restrict__ y_bfs,
    const unsigned short* __restrict__ Bw2s,
    const float* __restrict__ fc2_b, float* __restrict__ out) {
  __shared__ __align__(32) unsigned short As[8 * 4 * 512];
  __shared__ __align__(32) unsigned short Bs[8 * 4 * 512];
  const int tid = threadIdx.x, wave = tid >> 5, lane = tid & 31;
  const int m0 = blockIdx.x * 128, n0 = blockIdx.y * 128;
  {
    const char* Asrc = (const char*)(y_bfs + (size_t)blockIdx.x * (32 * 512));
    const char* Bsrc = (const char*)(Bw2s + (size_t)blockIdx.y * (32 * 512));
    unsigned ab = lds_off32(As), bb = lds_off32(Bs);
    for (int q = 0; q < 8; ++q) {
      int off = (q * 256 + tid) * 16;
      async_cp16(ab + off, Asrc + off);
      async_cp16(bb + off, Bsrc + off);
    }
  }
  wait_async();
  __syncthreads();
  v8f z = {};
  v8f acc[8];
  for (int i = 0; i < 8; ++i) acc[i] = z;
  for (int kt = 0; kt < 4; ++kt) {
    v16bf a = frag_ld(As + (wave * 4 + kt) * 512, lane);
    for (int nt = 0; nt < 8; ++nt)
      acc[nt] = wmma_bf16(a, frag_ld(Bs + (nt * 4 + kt) * 512, lane), acc[nt]);
  }
  int rbase = m0 + wave * 16 + ((lane >> 4) << 3);
  for (int nt = 0; nt < 8; ++nt) {
    int col = n0 + nt * 16 + (lane & 15);
    float b = fc2_b[col];
    for (int j = 0; j < 8; ++j)
      out[(size_t)(rbase + j) * DIN + col] = acc[nt][j] + b;
  }
}

// ---------------------------------------------------------------------------
extern "C" void kernel_launch(void* const* d_in, const int* in_sizes, int n_in,
                              void* d_out, int out_size, void* d_ws,
                              size_t ws_size, hipStream_t stream) {
  const float* x_s1  = (const float*)d_in[0];
  const float* x_att = (const float*)d_in[1];
  const float* fc_w  = (const float*)d_in[2];
  const float* fc_b  = (const float*)d_in[3];
  const float* fc2_w = (const float*)d_in[4];
  const float* fc2_b = (const float*)d_in[5];
  const float* w12   = (const float*)d_in[6];
  const float* w21   = (const float*)d_in[7];
  float* out = (float*)d_out;

  unsigned char* ws = (unsigned char*)d_ws;
  size_t off = 0;
  auto alloc = [&](size_t bytes) -> void* {
    void* p = ws + off;
    off += (bytes + 255) & ~(size_t)255;
    return p;
  };
  unsigned short* att_bf = (unsigned short*)alloc((size_t)ROWS * DHP * 2);
  unsigned short* c_bfs  = (unsigned short*)alloc((size_t)NBATCH * 4096 * 2);
  unsigned short* y_bfs  = (unsigned short*)alloc((size_t)ROWS * DHP * 2);
  unsigned short* Bw1s   = (unsigned short*)alloc((size_t)DIN * DHP * 2);
  unsigned short* Bw2s   = (unsigned short*)alloc((size_t)DHP * DIN * 2);
  unsigned short* B12s   = (unsigned short*)alloc((size_t)DHP * DHP * 2);
  unsigned short* B21s   = (unsigned short*)alloc((size_t)DHP * DHP * 2);

  int prep_elems = 2 * DIN * DHP + DHP * DHP;
  k_prep<<<(prep_elems + 255) / 256, 256, 0, stream>>>(fc_w, fc2_w, w12, w21,
                                                       Bw1s, Bw2s, B12s, B21s);
  k_gemm1<<<ROWS / 128, 256, 0, stream>>>(x_att, Bw1s, fc_b, att_bf);
  k_gemmc<<<NBATCH, 256, 0, stream>>>(att_bf, B12s, B21s, c_bfs);
  k_attn<<<NB2, 256, 0, stream>>>(att_bf, c_bfs, x_s1, y_bfs);
  k_gemm2<<<dim3(ROWS / 128, DIN / 128), 256, 0, stream>>>(y_bfs, Bw2s, fc2_b, out);
}
